// MaxViT_14843406975554
// MI455X (gfx1250) — matrix operand order repaired and checked by reading, not compiled, not run on hardware
//
#include <hip/hip_runtime.h>
#include <hip/hip_bf16.h>
#include <math.h>
#include <stdint.h>

// ---------------------------------------------------------------------------
// MaxViT forward for MI455X (gfx1250): bf16 WMMA GEMM pipeline.
// Branch-free 128x64 GEMM tiles (all problem dims divide the tile),
// vectorized b128 global<->LDS movement, 4 WMMAs per wave per K-step.
// ---------------------------------------------------------------------------

typedef __bf16 bf16_t;
typedef __attribute__((ext_vector_type(16))) __bf16 v16bf;
typedef __attribute__((ext_vector_type(8)))  __bf16 v8bf;
typedef __attribute__((ext_vector_type(8)))  float  v8f;

#define WH_   6
#define WW_   10
#define NWIN_ 60

// LDS strides (bf16 elements); rows are 16-byte aligned.
#define LDA_ 48
#define LDB_ 48

__device__ __forceinline__ v8f wmma_bf16(v16bf a, v16bf b, v8f c) {
  return __builtin_amdgcn_wmma_f32_16x16x32_bf16(false, a, false, b,
                                                 (short)0, c, false, false);
}

__device__ __forceinline__ v16bf cat8(v8bf lo, v8bf hi) {
  return __builtin_shufflevector(lo, hi, 0, 1, 2, 3, 4, 5, 6, 7, 8, 9, 10, 11,
                                 12, 13, 14, 15);
}

__device__ __forceinline__ v8bf zero8() {
  v8bf z;
#pragma unroll
  for (int j = 0; j < 8; ++j) z[j] = (__bf16)0.0f;
  return z;
}

// A-fragment: 16x32 bf16 row-major, stride ld (elements, 16B-aligned rows).
// lane L: row m=L&15, K-base kb=(L>>4)*8; f[0..7]=k=kb.., f[8..15]=k=kb+16..
__device__ __forceinline__ v16bf frag_a(const bf16_t* base, int ld, int lane) {
  int m  = lane & 15;
  int kb = (lane >> 4) * 8;
  const bf16_t* p = base + m * ld + kb;
  v8bf lo = *(const v8bf*)(p);
  v8bf hi = *(const v8bf*)(p + 16);
  return cat8(lo, hi);
}

// B-fragment from TRANSPOSED tile [N][K] (stride ld): lane L column n=L&15,
// K-base kb=(L>>4)*8; f[0..7]=k=kb.., f[8..15]=k=kb+16..
__device__ __forceinline__ v16bf frag_bT(const bf16_t* baseT, int ld,
                                         int lane) {
  int n  = lane & 15;
  int kb = (lane >> 4) * 8;
  const bf16_t* p = baseT + n * ld + kb;
  v8bf lo = *(const v8bf*)(p);
  v8bf hi = *(const v8bf*)(p + 16);
  return cat8(lo, hi);
}

// token ordering: 0 = canonical (b,h,w), 1 = block windows, 2 = grid windows
__device__ __forceinline__ void tok2hw(int mode, int token, int H, int W,
                                       int& b, int& h, int& w) {
  if (mode == 0) {
    int hw = H * W;
    b = token / hw;
    int r = token % hw;
    h = r / W;
    w = r % W;
  } else {
    int nbh = H / WH_, nbw = W / WW_;
    int t   = token % NWIN_;
    int win = (token / NWIN_) % (nbh * nbw);
    b       = token / (NWIN_ * nbh * nbw);
    int i = t / WW_, j = t % WW_;
    int wh = win / nbw, ww = win % nbw;
    if (mode == 1) { h = wh * WH_ + i;  w = ww * WW_ + j;  }
    else           { h = i * nbh + wh;  w = j * nbw + ww;  }
  }
}

// ---------------------------------------------------------------------------
// Weight prep: fp32 [O][I] -> bf16 [O][Ipad]  (same layout, K padded to 32)
// ---------------------------------------------------------------------------
__global__ void k_wt(const float* __restrict__ w, bf16_t* __restrict__ o,
                     int O, int I, int Ipad) {
  int i = blockIdx.x * 256 + threadIdx.x;
  if (i >= O * Ipad) return;
  int r = i / Ipad, c = i % Ipad;
  o[i] = (c < I) ? (bf16_t)w[(size_t)r * I + c] : (bf16_t)0.0f;
}

// ---------------------------------------------------------------------------
// Generic GEMM: C[M,N] = A[M,K](bf16) @ W[N,K](bf16)^T  (+bias [+GELU->bf16])
// REQUIRES: M % 128 == 0, N % 64 == 0, K % 32 == 0 (true for all calls here).
// Block tile 128x64, 8 waves, each wave a 32x32 macro-tile (4 WMMAs/K-step).
// epi: 0 fp32, 1 +bias fp32, 2 +bias GELU bf16.
// ---------------------------------------------------------------------------
__global__ __launch_bounds__(256) void k_gemm(
    const bf16_t* __restrict__ A, const bf16_t* __restrict__ Bw,
    const float* __restrict__ bias, float* __restrict__ Cf,
    bf16_t* __restrict__ Cb, int M, int N, int K, int epi) {
  __shared__ __align__(16) bf16_t As[128][LDA_];
  __shared__ __align__(16) bf16_t BsT[64][LDB_];
  int tid  = threadIdx.x;
  int wave = tid >> 5, lane = tid & 31;
  int m0 = blockIdx.y * 128, n0 = blockIdx.x * 64;
  int wm = (wave & 3) * 32;     // wave macro-tile row
  int wn = (wave >> 2) * 32;    // wave macro-tile col
  int lr = tid >> 2;            // staging row 0..63
  int lc = (tid & 3) * 8;       // staging chunk 0/8/16/24
  v8f acc00 = {}, acc01 = {}, acc10 = {}, acc11 = {};

  for (int k0 = 0; k0 < K; k0 += 32) {
    // A tile 128x32 : two aligned 16B vectors per thread (branch-free)
    *(v8bf*)(&As[lr][lc]) =
        *(const v8bf*)(A + (size_t)(m0 + lr) * K + k0 + lc);
    *(v8bf*)(&As[lr + 64][lc]) =
        *(const v8bf*)(A + (size_t)(m0 + lr + 64) * K + k0 + lc);
    // B tile [N][K] 64x32 : one vector per thread
    *(v8bf*)(&BsT[lr][lc]) =
        *(const v8bf*)(Bw + (size_t)(n0 + lr) * K + k0 + lc);
    if (k0 + 32 < K)
      __builtin_prefetch(A + (size_t)(m0 + lr) * K + k0 + 32 + lc, 0, 3);
    __syncthreads();
    v16bf a0 = frag_a(&As[wm][0], LDA_, lane);
    v16bf a1 = frag_a(&As[wm + 16][0], LDA_, lane);
    v16bf b0 = frag_bT(&BsT[wn][0], LDB_, lane);
    v16bf b1 = frag_bT(&BsT[wn + 16][0], LDB_, lane);
    acc00 = wmma_bf16(a0, b0, acc00);
    acc01 = wmma_bf16(a0, b1, acc01);
    acc10 = wmma_bf16(a1, b0, acc10);
    acc11 = wmma_bf16(a1, b1, acc11);
    __syncthreads();
  }

#pragma unroll
  for (int mi = 0; mi < 2; ++mi) {
#pragma unroll
    for (int ni = 0; ni < 2; ++ni) {
      v8f acc = mi ? (ni ? acc11 : acc10) : (ni ? acc01 : acc00);
      int n  = n0 + wn + ni * 16 + (lane & 15);
      int mb = m0 + wm + mi * 16 + (lane >> 4) * 8;
      float bv = bias ? bias[n] : 0.0f;
#pragma unroll
      for (int v = 0; v < 8; ++v) {
        int m = mb + v;
        float val = acc[v] + bv;
        if (epi == 2) {
          val = 0.5f * val * (1.0f + erff(val * 0.70710678118654752f));
          Cb[(size_t)m * N + n] = (bf16_t)val;
        } else {
          Cf[(size_t)m * N + n] = val;
        }
      }
    }
  }
}

// ---------------------------------------------------------------------------
// Implicit-im2col conv GEMM: Y(NCHW fp32) = conv(X) + bias
// M = B*Ho*Wo (%128==0), N = Cout (%64==0), weights padded to Kpad%32==0.
// ---------------------------------------------------------------------------
__global__ __launch_bounds__(256) void k_conv(
    const float* __restrict__ X, const bf16_t* __restrict__ Bw,
    const float* __restrict__ bias, float* __restrict__ Y, int Bn, int Cin,
    int Hin, int Win, int Cout, int Ho, int Wo, int KH, int KW, int stride,
    int pad) {
  __shared__ __align__(16) bf16_t As[128][LDA_];
  __shared__ __align__(16) bf16_t BsT[64][LDB_];
  const int N = Cout;
  const int Kreal = Cin * KH * KW;
  const int Kpad  = (Kreal + 31) & ~31;
  int tid  = threadIdx.x;
  int wave = tid >> 5, lane = tid & 31;
  int m0 = blockIdx.y * 128, n0 = blockIdx.x * 64;
  int wm = (wave & 3) * 32;
  int wn = (wave >> 2) * 32;
  int lr = tid >> 2;
  int lc = (tid & 3) * 8;
  v8f acc00 = {}, acc01 = {}, acc10 = {}, acc11 = {};

  // decompose this thread's two gather rows once
  int gb[2], goh[2], gow[2];
#pragma unroll
  for (int rr = 0; rr < 2; ++rr) {
    int gm = m0 + lr + rr * 64;
    gb[rr] = gm / (Ho * Wo);
    int rm = gm % (Ho * Wo);
    goh[rr] = rm / Wo;
    gow[rr] = rm % Wo;
  }

  for (int k0 = 0; k0 < Kpad; k0 += 32) {
    // A: im2col gather (scalar, inherently strided)
#pragma unroll
    for (int rr = 0; rr < 2; ++rr) {
      v8bf av = zero8();
#pragma unroll
      for (int j = 0; j < 8; ++j) {
        int gk = k0 + lc + j;
        float v = 0.0f;
        if (gk < Kreal) {
          int ci = gk / (KH * KW);
          int rk = gk % (KH * KW);
          int kh = rk / KW, kw = rk % KW;
          int h = goh[rr] * stride - pad + kh;
          int w = gow[rr] * stride - pad + kw;
          if (h >= 0 && h < Hin && w >= 0 && w < Win)
            v = X[(((size_t)gb[rr] * Cin + ci) * Hin + h) * Win + w];
        }
        av[j] = (bf16_t)v;
      }
      *(v8bf*)(&As[lr + rr * 64][lc]) = av;
    }
    // B: padded weights [Cout][Kpad], contiguous 16B vector (branch-free)
    *(v8bf*)(&BsT[lr][lc]) =
        *(const v8bf*)(Bw + (size_t)(n0 + lr) * Kpad + k0 + lc);
    __syncthreads();
    v16bf a0 = frag_a(&As[wm][0], LDA_, lane);
    v16bf a1 = frag_a(&As[wm + 16][0], LDA_, lane);
    v16bf b0 = frag_bT(&BsT[wn][0], LDB_, lane);
    v16bf b1 = frag_bT(&BsT[wn + 16][0], LDB_, lane);
    acc00 = wmma_bf16(a0, b0, acc00);
    acc01 = wmma_bf16(a0, b1, acc01);
    acc10 = wmma_bf16(a1, b0, acc10);
    acc11 = wmma_bf16(a1, b1, acc11);
    __syncthreads();
  }

#pragma unroll
  for (int mi = 0; mi < 2; ++mi) {
#pragma unroll
    for (int ni = 0; ni < 2; ++ni) {
      v8f acc = mi ? (ni ? acc11 : acc10) : (ni ? acc01 : acc00);
      int n  = n0 + wn + ni * 16 + (lane & 15);
      int mb = m0 + wm + mi * 16 + (lane >> 4) * 8;
      float bv = bias[n];
#pragma unroll
      for (int v = 0; v < 8; ++v) {
        int m  = mb + v;
        int b  = m / (Ho * Wo);
        int rm = m % (Ho * Wo);
        int oh = rm / Wo, ow = rm % Wo;
        Y[(((size_t)b * Cout + n) * Ho + oh) * Wo + ow] = acc[v] + bv;
      }
    }
  }
}

// ---------------------------------------------------------------------------
// LayerNorm + gather into [T, C] bf16 token buffer (mode-ordered tokens)
// ---------------------------------------------------------------------------
__global__ __launch_bounds__(256) void k_ln(
    const float* __restrict__ x, const float* __restrict__ g,
    const float* __restrict__ be, bf16_t* __restrict__ out, int C, int H,
    int W, int mode) {
  __shared__ float s1[256], s2[256];
  int t = blockIdx.x;
  int b, h, w;
  tok2hw(mode, t, H, W, b, h, w);
  const float* base = x + (size_t)b * C * H * W + (size_t)h * W + w;
  size_t cs = (size_t)H * W;
  float sum = 0.f, sq = 0.f;
  for (int c = threadIdx.x; c < C; c += 256) {
    float v = base[c * cs];
    sum += v; sq += v * v;
  }
  s1[threadIdx.x] = sum; s2[threadIdx.x] = sq;
  __syncthreads();
  for (int o = 128; o > 0; o >>= 1) {
    if (threadIdx.x < o) {
      s1[threadIdx.x] += s1[threadIdx.x + o];
      s2[threadIdx.x] += s2[threadIdx.x + o];
    }
    __syncthreads();
  }
  float mean = s1[0] / (float)C;
  float var  = s2[0] / (float)C - mean * mean;
  float inv  = rsqrtf(var + 1e-5f);
  for (int c = threadIdx.x; c < C; c += 256) {
    float v = (base[c * cs] - mean) * inv * g[c] + be[c];
    out[(size_t)t * C + c] = (bf16_t)v;
  }
}

// ---------------------------------------------------------------------------
// Residual scatter-add: x(NCHW) += y[T,C] (mode-ordered tokens)
// ---------------------------------------------------------------------------
__global__ void k_scatter_add(float* __restrict__ x, const float* __restrict__ y,
                              int C, int H, int W, int mode, int total) {
  int i = blockIdx.x * 256 + threadIdx.x;
  if (i >= total) return;
  int t = i / C, c = i % C;
  int b, h, w;
  tok2hw(mode, t, H, W, b, h, w);
  x[(((size_t)b * C + c) * H + h) * W + w] += y[i];
}

// ---------------------------------------------------------------------------
// Fused window attention: one block per (window, head). 60 tokens -> pad 64.
// qkv fp32 [T, 3d]; rel-pos bias computed in-kernel. Output bf16 [T, d].
// ---------------------------------------------------------------------------
__global__ __launch_bounds__(256) void k_attn(
    const float* __restrict__ qkv, const float* __restrict__ relb,
    bf16_t* __restrict__ out, int heads, int d) {
  __shared__ __align__(16) bf16_t qs[64][LDA_];   // Q   [m][k]
  __shared__ __align__(16) bf16_t ks[64][LDA_];   // K^T [n][k] (token-major)
  __shared__ __align__(16) bf16_t vsT[32][64];    // V^T [n][k]
  __shared__ __align__(16) float  sim[64][72];
  __shared__ __align__(16) bf16_t ps[64][72];
  int head = blockIdx.x % heads;
  int win  = blockIdx.x / heads;
  int tid  = threadIdx.x;
  const float scale = 0.17677669529663687f;  // 32^-0.5

  // gather q/k/v: 64x32 each; thread loads 8 contiguous channels (float4 x2)
  {
    int t  = tid >> 2;
    int c0 = (tid & 3) * 8;
    float qv[8], kv[8], vv[8];
    if (t < NWIN_) {
      size_t row =
          ((size_t)win * NWIN_ + t) * (3 * d) + (size_t)head * 32 + c0;
      const float4* qp = (const float4*)(qkv + row);
      const float4* kp = (const float4*)(qkv + row + d);
      const float4* vp = (const float4*)(qkv + row + 2 * d);
      float4 q0 = qp[0], q1 = qp[1];
      float4 kk0 = kp[0], kk1 = kp[1];
      float4 vv0 = vp[0], vv1 = vp[1];
      qv[0]=q0.x; qv[1]=q0.y; qv[2]=q0.z; qv[3]=q0.w;
      qv[4]=q1.x; qv[5]=q1.y; qv[6]=q1.z; qv[7]=q1.w;
      kv[0]=kk0.x; kv[1]=kk0.y; kv[2]=kk0.z; kv[3]=kk0.w;
      kv[4]=kk1.x; kv[5]=kk1.y; kv[6]=kk1.z; kv[7]=kk1.w;
      vv[0]=vv0.x; vv[1]=vv0.y; vv[2]=vv0.z; vv[3]=vv0.w;
      vv[4]=vv1.x; vv[5]=vv1.y; vv[6]=vv1.z; vv[7]=vv1.w;
    } else {
#pragma unroll
      for (int j = 0; j < 8; ++j) { qv[j] = 0.f; kv[j] = 0.f; vv[j] = 0.f; }
    }
    v8bf qb, kb;
#pragma unroll
    for (int j = 0; j < 8; ++j) {
      qb[j] = (bf16_t)(qv[j] * scale);
      kb[j] = (bf16_t)kv[j];
    }
    *(v8bf*)(&qs[t][c0]) = qb;
    *(v8bf*)(&ks[t][c0]) = kb;
#pragma unroll
    for (int j = 0; j < 8; ++j) vsT[c0 + j][t] = (bf16_t)vv[j];  // transpose
  }
  __syncthreads();

  int wave = tid >> 5, lane = tid & 31;

  // scores: 16 tiles of 16x16 (K=32), two per wave
  for (int tile = wave; tile < 16; tile += 8) {
    int tm = tile >> 2, tn = tile & 3;
    v16bf a = frag_a(&qs[tm * 16][0], LDA_, lane);
    v16bf b = frag_bT(&ks[tn * 16][0], LDA_, lane);
    v8f c = {};
    c = wmma_bf16(a, b, c);
    int n  = tn * 16 + (lane & 15);
    int mb = tm * 16 + (lane >> 4) * 8;
#pragma unroll
    for (int v2 = 0; v2 < 8; ++v2) {
      int m = mb + v2;
      float s;
      if (n < NWIN_) {
        float bias = 0.0f;
        if (m < NWIN_) {
          int rel = ((m / WW_) - (n / WW_) + (WH_ - 1)) * (2 * WW_ - 1) +
                    ((m % WW_) - (n % WW_) + (WW_ - 1));
          bias = relb[rel * heads + head];
        }
        s = c[v2] + bias;
      } else {
        s = -1e30f;  // mask padded columns
      }
      sim[m][n] = s;
    }
  }
  __syncthreads();

  // row softmax
  if (tid < 64) {
    float mx = -1e30f;
    for (int j = 0; j < 64; ++j) mx = fmaxf(mx, sim[tid][j]);
    float ssum = 0.f;
    for (int j = 0; j < 64; ++j) {
      float e = __expf(sim[tid][j] - mx);
      sim[tid][j] = e;
      ssum += e;
    }
    float r = 1.0f / ssum;
    for (int j = 0; j < 64; ++j) ps[tid][j] = (bf16_t)(sim[tid][j] * r);
  }
  __syncthreads();

  // out = P(64x64) @ V(64x32): 8 tiles, one per wave, K = 64 (2 x 32)
  {
    int tm = wave >> 1, tn = wave & 1;
    v8f c = {};
#pragma unroll
    for (int ks0 = 0; ks0 < 64; ks0 += 32) {
      v16bf a = frag_a(&ps[tm * 16][ks0], 72, lane);
      v16bf b = frag_bT(&vsT[tn * 16][ks0], 64, lane);
      c = wmma_bf16(a, b, c);
    }
    int n  = tn * 16 + (lane & 15);
    int mb = tm * 16 + (lane >> 4) * 8;
#pragma unroll
    for (int v2 = 0; v2 < 8; ++v2) {
      int m = mb + v2;
      if (m < NWIN_)
        out[((size_t)win * NWIN_ + m) * d + (size_t)head * 32 + n] =
            (bf16_t)c[v2];
    }
  }
}

// ---------------------------------------------------------------------------
// Host orchestration
// ---------------------------------------------------------------------------
extern "C" void kernel_launch(void* const* d_in, const int* in_sizes, int n_in,
                              void* d_out, int out_size, void* d_ws,
                              size_t ws_size, hipStream_t stream) {
  (void)in_sizes; (void)n_in; (void)ws_size;

  // ---- parse inputs (setup_inputs order) ----
  int idx = 0;
  const float* X = (const float*)d_in[idx++];
  struct AttnP { const float *lng, *lnb, *wqkv, *wout, *rel; };
  struct FfnP  { const float *lng, *lnb, *w1, *b1, *w2, *b2; };
  struct Stage {
    const float *conv_w, *conv_b;
    AttnP a1, a2; FfnP f1, f2;
  } st[4];
  for (int i = 0; i < 4; ++i) {
    st[i].conv_w = (const float*)d_in[idx++];
    st[i].conv_b = (const float*)d_in[idx++];
    st[i].a1 = { (const float*)d_in[idx], (const float*)d_in[idx+1],
                 (const float*)d_in[idx+2], (const float*)d_in[idx+3],
                 (const float*)d_in[idx+4] }; idx += 5;
    st[i].f1 = { (const float*)d_in[idx], (const float*)d_in[idx+1],
                 (const float*)d_in[idx+2], (const float*)d_in[idx+3],
                 (const float*)d_in[idx+4], (const float*)d_in[idx+5] }; idx += 6;
    st[i].a2 = { (const float*)d_in[idx], (const float*)d_in[idx+1],
                 (const float*)d_in[idx+2], (const float*)d_in[idx+3],
                 (const float*)d_in[idx+4] }; idx += 5;
    st[i].f2 = { (const float*)d_in[idx], (const float*)d_in[idx+1],
                 (const float*)d_in[idx+2], (const float*)d_in[idx+3],
                 (const float*)d_in[idx+4], (const float*)d_in[idx+5] }; idx += 6;
  }

  // ---- workspace layout (all regions 1MB aligned) ----
  char* ws = (char*)d_ws;
  float*  actA = (float*)(ws);                               // 32 MB
  float*  actB = (float*)(ws + (size_t)32  * (1u << 20));    // 32 MB
  bf16_t* tokA = (bf16_t*)(ws + (size_t)64  * (1u << 20));   // 16 MB
  bf16_t* tokB = (bf16_t*)(ws + (size_t)80  * (1u << 20));   // 64 MB
  float*  fbuf = (float*)(ws + (size_t)144 * (1u << 20));    // 96 MB
  bf16_t* wts  = (bf16_t*)(ws + (size_t)240 * (1u << 20));   // 24 MB

  size_t wcur = 0;
  auto prep_wt = [&](const float* w, int O, int I, int Ipad) -> const bf16_t* {
    bf16_t* p = wts + wcur;
    wcur += (size_t)O * Ipad;
    wcur = (wcur + 15) & ~(size_t)15;
    int tot = O * Ipad;
    k_wt<<<(tot + 255) / 256, 256, 0, stream>>>(w, p, O, I, Ipad);
    return p;
  };

  auto launch_gemm = [&](const bf16_t* A, const bf16_t* Bw, const float* bias,
                         float* Cf, bf16_t* Cb, int M, int N, int K, int epi) {
    dim3 g(N / 64, M / 128);
    k_gemm<<<g, 256, 0, stream>>>(A, Bw, bias, Cf, Cb, M, N, K, epi);
  };

  const int CinA[4]  = {20, 64, 128, 256};
  const int CoutA[4] = {64, 128, 256, 512};
  const int HinA[4]  = {384, 96, 48, 24};
  const int WinA[4]  = {640, 160, 80, 40};
  const int HoA[4]   = {96, 48, 24, 12};
  const int WoA[4]   = {160, 80, 40, 20};
  const int KKA[4]   = {7, 3, 3, 3};
  const int STA[4]   = {4, 2, 2, 2};
  const int PDA[4]   = {2, 1, 1, 1};

  float* bufs[2] = {actA, actB};

  for (int i = 0; i < 4; ++i) {
    const int Cin = CinA[i], Cout = CoutA[i];
    const int Hin = HinA[i], Win = WinA[i];
    const int Ho = HoA[i], Wo = WoA[i];
    const int d = Cout, H = Ho, W = Wo;
    const int T = 8 * H * W;
    const int heads = d / 32;
    const int nWinTot = T / NWIN_;

    const int Kc    = Cin * KKA[i] * KKA[i];
    const int KcPad = (Kc + 31) & ~31;

    // bf16 weights for this stage ([O][Ipad], no transpose needed)
    const bf16_t* convw = prep_wt(st[i].conv_w, Cout, Kc, KcPad);
    const bf16_t* wqkv1 = prep_wt(st[i].a1.wqkv, 3 * d, d, d);
    const bf16_t* wout1 = prep_wt(st[i].a1.wout, d, d, d);
    const bf16_t* w11   = prep_wt(st[i].f1.w1, 4 * d, d, d);
    const bf16_t* w21   = prep_wt(st[i].f1.w2, d, 4 * d, 4 * d);
    const bf16_t* wqkv2 = prep_wt(st[i].a2.wqkv, 3 * d, d, d);
    const bf16_t* wout2 = prep_wt(st[i].a2.wout, d, d, d);
    const bf16_t* w12   = prep_wt(st[i].f2.w1, 4 * d, d, d);
    const bf16_t* w22   = prep_wt(st[i].f2.w2, d, 4 * d, 4 * d);

    // conv
    const float* src = (i == 0) ? X : bufs[(i - 1) & 1];
    float* xact = bufs[i & 1];
    {
      int M = 8 * Ho * Wo, N = Cout;
      dim3 g(N / 64, M / 128);
      k_conv<<<g, 256, 0, stream>>>(src, convw, st[i].conv_b, xact, 8, Cin,
                                    Hin, Win, Cout, Ho, Wo, KKA[i], KKA[i],
                                    STA[i], PDA[i]);
    }

    auto attn_block = [&](const AttnP& p, const bf16_t* wqkv,
                          const bf16_t* wout, int mode) {
      k_ln<<<T, 256, 0, stream>>>(xact, p.lng, p.lnb, tokA, d, H, W, mode);
      launch_gemm(tokA, wqkv, nullptr, fbuf, nullptr, T, 3 * d, d, 0);
      k_attn<<<nWinTot * heads, 256, 0, stream>>>(fbuf, p.rel, tokA, heads, d);
      launch_gemm(tokA, wout, nullptr, fbuf, nullptr, T, d, d, 0);
      k_scatter_add<<<((size_t)T * d + 255) / 256, 256, 0, stream>>>(
          xact, fbuf, d, H, W, mode, T * d);
    };
    auto ffn_block = [&](const FfnP& p, const bf16_t* w1b, const bf16_t* w2b) {
      k_ln<<<T, 256, 0, stream>>>(xact, p.lng, p.lnb, tokA, d, H, W, 0);
      launch_gemm(tokA, w1b, p.b1, nullptr, tokB, T, 4 * d, d, 2);   // GELU->bf16
      launch_gemm(tokB, w2b, p.b2, fbuf, nullptr, T, d, 4 * d, 1);   // +bias fp32
      k_scatter_add<<<((size_t)T * d + 255) / 256, 256, 0, stream>>>(
          xact, fbuf, d, H, W, 0, T * d);
    };

    attn_block(st[i].a1, wqkv1, wout1, /*block windows*/ 1);
    ffn_block(st[i].f1, w11, w21);
    attn_block(st[i].a2, wqkv2, wout2, /*grid windows*/ 2);
    ffn_block(st[i].f2, w12, w22);
  }

  // final activations live in bufs[3 & 1] == actB, canonical [8,512,12,20]
  hipMemcpyAsync(d_out, bufs[3 & 1], (size_t)out_size * sizeof(float),
                 hipMemcpyDeviceToDevice, stream);
}
